// GATModel_21449066676756
// MI455X (gfx1250) — compile-verified, matrix operand order checked by reading, not run on hardware
//
#include <hip/hip_runtime.h>
#include <hip/hip_bf16.h>
#include <math.h>

typedef __attribute__((ext_vector_type(16))) _Float16 v16h;
typedef __attribute__((ext_vector_type(8)))  float    v8f;

#define NEG_SLOPE 0.2f

// ============================================================================
// B-matrix pre-pack: W[K,N] f32 (row major) -> f16 fragments in WMMA B layout.
// One wave per (tileN, kBlock). Output layout:
//   P[((tn*nKB + kb)*32 + lane)*16 + j]  (16 f16 = 32B contiguous per lane)
// B (32x16 f16) fragment layout per CDNA5 ISA 7.12.2: lane L -> col N=L&15;
// elems 0..7 hold K=(L>>4)*8+j, elems 8..15 hold K=16+(L>>4)*8+(j-8).
// Columns >= N are zero-filled (used for the N=1 layer-2 GEMMs).
// ============================================================================
__global__ __launch_bounds__(32)
void gat_pack_b(const float* __restrict__ W, _Float16* __restrict__ P,
                int K, int N, int nKB)
{
    const int tn   = blockIdx.x;
    const int kb   = blockIdx.y;
    const int lane = threadIdx.x & 31;
    const int hsel = lane >> 4;
    const int col  = tn * 16 + (lane & 15);
    const bool cOK = col < N;

    v16h frag;
#pragma unroll
    for (int j = 0; j < 16; ++j) {
        const int kk = kb * 32 + hsel * 8 + ((j < 8) ? j : (j + 8)); // < K by grid
        frag[j] = (_Float16)(cOK ? W[(size_t)kk * N + col] : 0.0f);
    }
    *(v16h*)(P + ((size_t)(tn * nKB + kb) * 32 + lane) * 16) = frag;
}

// ============================================================================
// WMMA GEMM: Y[M,N] = act(X[M,K] @ W[K,N] (+ bias[N]))
// One wave per 16x16 tile. K is a compile-time multiple of 32 -> the K loop
// fully unrolls into K/32 back-to-back v_wmma_f32_16x16x32_f16.
// A loads: branchless, row index clamped (garbage rows never stored), four
// aligned float4 (global_load_b128) per K-step. B loads: one 32B v16h per
// lane from the pre-packed fragment buffer.
// Store epilogue: single cOK exec-mask + hoisted full-tile fast path (M is a
// multiple of 16 at every call site), 8 straight-line stores off one base.
// ============================================================================
template <int K>
__global__ __launch_bounds__(32)
void gat_gemm_wmma(const float* __restrict__ X, const _Float16* __restrict__ BP,
                   const float* __restrict__ bias, float* __restrict__ Y,
                   int M, int N, int do_relu)
{
    constexpr int nKB = K / 32;
    const int lane = threadIdx.x & 31;
    const int hsel = lane >> 4;
    const int l16  = lane & 15;

    int rowA = blockIdx.x * 16 + l16;
    rowA = (rowA < M) ? rowA : (M - 1);            // clamp: loads stay in bounds
    const float* __restrict__ xr = X + (size_t)rowA * K;
    const int colB = blockIdx.y * 16 + l16;
    const bool cOK = colB < N;
    const _Float16* __restrict__ bp =
        BP + ((size_t)(blockIdx.y * nKB) * 32 + lane) * 16;

    v8f acc = {};
#pragma unroll
    for (int kb = 0; kb < nKB; ++kb) {
        const int k0 = kb * 32;
        const float4 x0 = *(const float4*)(xr + k0 + hsel * 8);
        const float4 x1 = *(const float4*)(xr + k0 + hsel * 8 + 4);
        const float4 x2 = *(const float4*)(xr + k0 + 16 + hsel * 8);
        const float4 x3 = *(const float4*)(xr + k0 + 16 + hsel * 8 + 4);
        v16h a;
        a[0]  = (_Float16)x0.x; a[1]  = (_Float16)x0.y;
        a[2]  = (_Float16)x0.z; a[3]  = (_Float16)x0.w;
        a[4]  = (_Float16)x1.x; a[5]  = (_Float16)x1.y;
        a[6]  = (_Float16)x1.z; a[7]  = (_Float16)x1.w;
        a[8]  = (_Float16)x2.x; a[9]  = (_Float16)x2.y;
        a[10] = (_Float16)x2.z; a[11] = (_Float16)x2.w;
        a[12] = (_Float16)x3.x; a[13] = (_Float16)x3.y;
        a[14] = (_Float16)x3.z; a[15] = (_Float16)x3.w;

        const v16h b = *(const v16h*)(bp + (size_t)kb * 32 * 16);

        acc = __builtin_amdgcn_wmma_f32_16x16x32_f16(
            /*neg_a=*/false, a, /*neg_b=*/false, b,
            /*c_mod=*/(short)0, acc, /*reuse_a=*/false, /*reuse_b=*/false);
    }

    if (cOK) {
        const float bv = bias ? bias[colB] : 0.0f;
        const int rbase = blockIdx.x * 16 + hsel * 8;
        float* __restrict__ yp = Y + (size_t)rbase * N + colB;
        if (rbase + 8 <= M) {                       // always at runtime here
#pragma unroll
            for (int j = 0; j < 8; ++j) {
                float v = acc[j] + bv;
                if (do_relu) v = fmaxf(v, 0.0f);
                yp[(size_t)j * N] = v;
            }
        } else {
#pragma unroll
            for (int j = 0; j < 8; ++j) {
                if (rbase + j < M) {
                    float v = acc[j] + bv;
                    if (do_relu) v = fmaxf(v, 0.0f);
                    yp[(size_t)j * N] = v;
                }
            }
        }
    }
}

// ============================================================================
// Elementwise / edge kernels (H, O, F are compile-time at every call site)
// ============================================================================
__global__ void gat_fill(float* __restrict__ p, float val, size_t n)
{
    size_t i = (size_t)blockIdx.x * blockDim.x + threadIdx.x;
    if (i < n) p[i] = val;
}

// out[n,h] = sum_o F[n,h,o] * a[h,o]
template <int H, int O>
__global__ void gat_attn_score(const float* __restrict__ F, const float* __restrict__ a,
                               float* __restrict__ out, int N)
{
    int idx = blockIdx.x * blockDim.x + threadIdx.x;
    if (idx >= N * H) return;
    const int n = idx / H, h = idx % H;        // H power of two -> shift/mask
    const float* f  = F + (size_t)n * H * O + (size_t)h * O;
    const float* av = a + (size_t)h * O;
    float s = 0.0f;
#pragma unroll
    for (int o = 0; o < O; ++o) s += f[o] * av[o];
    out[idx] = s;
}

__device__ __forceinline__ void atomicMaxF(float* addr, float val)
{
    unsigned int* ai = (unsigned int*)addr;
    unsigned int old = *ai;
    while (true) {
        float cur = __uint_as_float(old);
        if (cur >= val) break;
        unsigned int assumed = old;
        old = atomicCAS(ai, assumed, __float_as_uint(val));
        if (old == assumed) break;
    }
}

// e[e,h] = leaky_relu(el[src,h] + er[dst,h]); m[dst,h] = max(...)
template <int H>
__global__ void gat_edge_max(const float* __restrict__ el, const float* __restrict__ er,
                             const int* __restrict__ src, const int* __restrict__ dst,
                             float* __restrict__ e_out, float* __restrict__ m,
                             int E)
{
    int idx = blockIdx.x * blockDim.x + threadIdx.x;
    if (idx >= E * H) return;
    const int e = idx / H, h = idx % H;
    const int s = src[e], d = dst[e];
    float v = el[(size_t)s * H + h] + er[(size_t)d * H + h];
    v = (v > 0.0f) ? v : NEG_SLOPE * v;
    e_out[idx] = v;
    atomicMaxF(&m[(size_t)d * H + h], v);
}

// in place: e[e,h] = exp(e[e,h] - m[dst,h]); denom[dst,h] += e
template <int H>
__global__ void gat_edge_exp(float* __restrict__ e_io, const float* __restrict__ m,
                             const int* __restrict__ dst, float* __restrict__ denom,
                             int E)
{
    int idx = blockIdx.x * blockDim.x + threadIdx.x;
    if (idx >= E * H) return;
    const int e = idx / H, h = idx % H;
    const int d = dst[e];
    float mv = m[(size_t)d * H + h];
    if (!isfinite(mv)) mv = 0.0f;           // isolated-node convention (ref)
    const float ex = __expf(e_io[idx] - mv);
    e_io[idx] = ex;
    atomicAdd(&denom[(size_t)d * H + h], ex);
}

// Layer-1 scatter, specialized F=128 (H=4, O=32): one float4 per thread.
__global__ void gat_edge_aggr128(const float* __restrict__ ex, const float* __restrict__ denom,
                                 const float* __restrict__ fs,
                                 const int* __restrict__ src, const int* __restrict__ dst,
                                 float* __restrict__ agg, int E)
{
    long long idx = (long long)blockIdx.x * blockDim.x + threadIdx.x;
    if (idx >= (long long)E * 32) return;
    const int e = (int)(idx >> 5);
    const int q = (int)(idx & 31);          // which float4 (feature group)
    const int f = q * 4;
    const int h = q >> 3;                   // 32 features per head
    const int s = src[e], d = dst[e];
    const float alpha = ex[(size_t)e * 4 + h] /
                        fmaxf(denom[(size_t)d * 4 + h], 1e-9f);
    const float4 v = *(const float4*)(fs + (size_t)s * 128 + f);
    float* ap = agg + (size_t)d * 128 + f;
    atomicAdd(ap + 0, alpha * v.x);
    atomicAdd(ap + 1, alpha * v.y);
    atomicAdd(ap + 2, alpha * v.z);
    atomicAdd(ap + 3, alpha * v.w);
}

// Layer-2 scatter (H=1, O=1): one scalar per edge.
__global__ void gat_edge_aggr1(const float* __restrict__ ex, const float* __restrict__ denom,
                               const float* __restrict__ fs,
                               const int* __restrict__ src, const int* __restrict__ dst,
                               float* __restrict__ agg, int E)
{
    int e = blockIdx.x * blockDim.x + threadIdx.x;
    if (e >= E) return;
    const int s = src[e], d = dst[e];
    const float alpha = ex[e] / fmaxf(denom[d], 1e-9f);
    atomicAdd(&agg[d], alpha * fs[s]);
}

// in place: A[n,f] = relu(A[n,f] + b[f]),  F = 128 (power of two)
template <int F>
__global__ void gat_bias_relu(float* __restrict__ A, const float* __restrict__ b,
                              size_t total)
{
    size_t i = (size_t)blockIdx.x * blockDim.x + threadIdx.x;
    if (i < total) A[i] = fmaxf(A[i] + b[i & (F - 1)], 0.0f);
}

// out[n] = relu(agg[n] + b[0])
__global__ void gat_out(const float* __restrict__ agg, const float* __restrict__ b,
                        float* __restrict__ out, int N)
{
    int i = blockIdx.x * blockDim.x + threadIdx.x;
    if (i < N) out[i] = fmaxf(agg[i] + b[0], 0.0f);
}

// ============================================================================
// Orchestration
// ============================================================================
static inline unsigned cdiv(long long a, long long b) { return (unsigned)((a + b - 1) / b); }

extern "C" void kernel_launch(void* const* d_in, const int* in_sizes, int n_in,
                              void* d_out, int out_size, void* d_ws, size_t ws_size,
                              hipStream_t stream)
{
    const float* feat_user = (const float*)d_in[0];
    const float* feat_item = (const float*)d_in[1];
    const float* Wp_u  = (const float*)d_in[2];
    const float* Wp_i  = (const float*)d_in[3];
    const float* W1_ui = (const float*)d_in[4];
    const float* al1_ui= (const float*)d_in[5];
    const float* ar1_ui= (const float*)d_in[6];
    const float* b1_ui = (const float*)d_in[7];
    const float* W1_iu = (const float*)d_in[8];
    const float* al1_iu= (const float*)d_in[9];
    const float* ar1_iu= (const float*)d_in[10];
    const float* b1_iu = (const float*)d_in[11];
    const float* W2_ui = (const float*)d_in[12];
    const float* al2_ui= (const float*)d_in[13];
    const float* ar2_ui= (const float*)d_in[14];
    const float* b2_ui = (const float*)d_in[15];
    const float* W2_iu = (const float*)d_in[16];
    const float* al2_iu= (const float*)d_in[17];
    const float* ar2_iu= (const float*)d_in[18];
    const float* b2_iu = (const float*)d_in[19];
    const int* src_ui = (const int*)d_in[20];
    const int* dst_ui = (const int*)d_in[21];
    const int* src_iu = (const int*)d_in[22];
    const int* dst_iu = (const int*)d_in[23];

    const int Nu = in_sizes[0] / 128;
    const int Ni = in_sizes[1] / 64;
    const int E  = in_sizes[20];

    // ---- workspace bump allocator (256B aligned) ----
    char* wsb = (char*)d_ws;
    size_t off = 0;
    auto allocB = [&](size_t bytes) -> void* {
        void* p = (void*)(wsb + off);
        off += (bytes + 255) & ~(size_t)255;
        return p;
    };
    auto allocF = [&](size_t elems) -> float*     { return (float*)allocB(elems * 4); };
    auto allocH = [&](size_t elems) -> _Float16*  { return (_Float16*)allocB(elems * 2); };

    float* hu     = allocF((size_t)Nu * 32);
    float* hi     = allocF((size_t)Ni * 32);
    float* fs_ui  = allocF((size_t)Nu * 128);
    float* fd_ui  = allocF((size_t)Ni * 128);
    float* fs_iu  = allocF((size_t)Ni * 128);
    float* fd_iu  = allocF((size_t)Nu * 128);
    float* el_ui  = allocF((size_t)Nu * 4);
    float* er_ui  = allocF((size_t)Ni * 4);
    float* el_iu  = allocF((size_t)Ni * 4);
    float* er_iu  = allocF((size_t)Nu * 4);
    float* m_ui   = allocF((size_t)Ni * 4);
    float* dn_ui  = allocF((size_t)Ni * 4);
    float* m_iu   = allocF((size_t)Nu * 4);
    float* dn_iu  = allocF((size_t)Nu * 4);
    float* e_ui   = allocF((size_t)E * 4);
    float* e_iu   = allocF((size_t)E * 4);
    float* agg_ui = allocF((size_t)Ni * 128);   // -> item hidden (in place)
    float* agg_iu = allocF((size_t)Nu * 128);   // -> user hidden (in place)
    float* fs2_ui = allocF((size_t)Nu);
    float* fd2_ui = allocF((size_t)Ni);
    float* fs2_iu = allocF((size_t)Ni);
    float* fd2_iu = allocF((size_t)Nu);
    float* el2_ui = allocF((size_t)Nu);
    float* er2_ui = allocF((size_t)Ni);
    float* el2_iu = allocF((size_t)Ni);
    float* er2_iu = allocF((size_t)Nu);
    float* m2_ui  = allocF((size_t)Ni);
    float* dn2_ui = allocF((size_t)Ni);
    float* m2_iu  = allocF((size_t)Nu);
    float* dn2_iu = allocF((size_t)Nu);
    float* e2_ui  = allocF((size_t)E);
    float* e2_iu  = allocF((size_t)E);
    float* agg2_ui= allocF((size_t)Ni);
    float* agg2_iu= allocF((size_t)Nu);
    // packed f16 B-fragment buffers (tiny)
    _Float16* pWp_u  = allocH(2 * 4 * 512);  // K=128,N=32 : nTN=2, nKB=4
    _Float16* pWp_i  = allocH(2 * 2 * 512);  // K=64, N=32 : nTN=2, nKB=2
    _Float16* pW1_ui = allocH(8 * 1 * 512);  // K=32, N=128: nTN=8, nKB=1
    _Float16* pW1_iu = allocH(8 * 1 * 512);
    _Float16* pW2_ui = allocH(1 * 4 * 512);  // K=128,N=1 : nTN=1, nKB=4
    _Float16* pW2_iu = allocH(1 * 4 * 512);
    (void)ws_size; (void)n_in; (void)out_size;

    const int TB = 256;
    auto packB = [&](const float* W, _Float16* P, int K, int N) {
        dim3 g(cdiv(N, 16), K / 32);
        gat_pack_b<<<g, 32, 0, stream>>>(W, P, K, N, K / 32);
    };
    auto gemm = [&](const float* X, const _Float16* BP, const float* bias, float* Y,
                    int M, int K, int N, int relu) {
        dim3 g(cdiv(M, 16), cdiv(N, 16));
        switch (K) {
            case 32:  gat_gemm_wmma<32><<<g, 32, 0, stream>>>(X, BP, bias, Y, M, N, relu); break;
            case 64:  gat_gemm_wmma<64><<<g, 32, 0, stream>>>(X, BP, bias, Y, M, N, relu); break;
            default:  gat_gemm_wmma<128><<<g, 32, 0, stream>>>(X, BP, bias, Y, M, N, relu); break;
        }
    };

    // ---- pack all weight matrices into WMMA B-fragment layout ----
    packB(Wp_u,  pWp_u,  128, 32);
    packB(Wp_i,  pWp_i,   64, 32);
    packB(W1_ui, pW1_ui,  32, 128);
    packB(W1_iu, pW1_iu,  32, 128);
    packB(W2_ui, pW2_ui, 128, 1);
    packB(W2_iu, pW2_iu, 128, 1);

    // ---- input projections (WMMA) ----
    gemm(feat_user, pWp_u, nullptr, hu, Nu, 128, 32, 1);
    gemm(feat_item, pWp_i, nullptr, hi, Ni,  64, 32, 1);

    // ---- layer-1 feature transforms (WMMA) ----
    gemm(hu, pW1_ui, nullptr, fs_ui, Nu, 32, 128, 0);
    gemm(hi, pW1_ui, nullptr, fd_ui, Ni, 32, 128, 0);
    gemm(hi, pW1_iu, nullptr, fs_iu, Ni, 32, 128, 0);
    gemm(hu, pW1_iu, nullptr, fd_iu, Nu, 32, 128, 0);

    // ---- layer-1 attention scores ----
    gat_attn_score<4, 32><<<cdiv((long long)Nu * 4, TB), TB, 0, stream>>>(fs_ui, al1_ui, el_ui, Nu);
    gat_attn_score<4, 32><<<cdiv((long long)Ni * 4, TB), TB, 0, stream>>>(fd_ui, ar1_ui, er_ui, Ni);
    gat_attn_score<4, 32><<<cdiv((long long)Ni * 4, TB), TB, 0, stream>>>(fs_iu, al1_iu, el_iu, Ni);
    gat_attn_score<4, 32><<<cdiv((long long)Nu * 4, TB), TB, 0, stream>>>(fd_iu, ar1_iu, er_iu, Nu);

    // ---- layer-1 softmax + aggregate : relation ui (dst = items) ----
    gat_fill<<<cdiv((long long)Ni * 4, TB), TB, 0, stream>>>(m_ui, -INFINITY, (size_t)Ni * 4);
    gat_fill<<<cdiv((long long)Ni * 4, TB), TB, 0, stream>>>(dn_ui, 0.0f, (size_t)Ni * 4);
    gat_fill<<<cdiv((long long)Ni * 128, TB), TB, 0, stream>>>(agg_ui, 0.0f, (size_t)Ni * 128);
    gat_edge_max<4><<<cdiv((long long)E * 4, TB), TB, 0, stream>>>(el_ui, er_ui, src_ui, dst_ui, e_ui, m_ui, E);
    gat_edge_exp<4><<<cdiv((long long)E * 4, TB), TB, 0, stream>>>(e_ui, m_ui, dst_ui, dn_ui, E);
    gat_edge_aggr128<<<cdiv((long long)E * 32, TB), TB, 0, stream>>>(e_ui, dn_ui, fs_ui, src_ui, dst_ui, agg_ui, E);

    // ---- layer-1 softmax + aggregate : relation iu (dst = users) ----
    gat_fill<<<cdiv((long long)Nu * 4, TB), TB, 0, stream>>>(m_iu, -INFINITY, (size_t)Nu * 4);
    gat_fill<<<cdiv((long long)Nu * 4, TB), TB, 0, stream>>>(dn_iu, 0.0f, (size_t)Nu * 4);
    gat_fill<<<cdiv((long long)Nu * 128, TB), TB, 0, stream>>>(agg_iu, 0.0f, (size_t)Nu * 128);
    gat_edge_max<4><<<cdiv((long long)E * 4, TB), TB, 0, stream>>>(el_iu, er_iu, src_iu, dst_iu, e_iu, m_iu, E);
    gat_edge_exp<4><<<cdiv((long long)E * 4, TB), TB, 0, stream>>>(e_iu, m_iu, dst_iu, dn_iu, E);
    gat_edge_aggr128<<<cdiv((long long)E * 32, TB), TB, 0, stream>>>(e_iu, dn_iu, fs_iu, src_iu, dst_iu, agg_iu, E);

    // ---- bias + relu -> layer-2 hidden (in place) ----
    gat_bias_relu<128><<<cdiv((long long)Nu * 128, TB), TB, 0, stream>>>(agg_iu, b1_iu, (size_t)Nu * 128); // users
    gat_bias_relu<128><<<cdiv((long long)Ni * 128, TB), TB, 0, stream>>>(agg_ui, b1_ui, (size_t)Ni * 128); // items
    float* h2u = agg_iu;
    float* h2i = agg_ui;

    // ---- layer-2 transforms (WMMA, N=1 via zero-padded B fragments) ----
    gemm(h2u, pW2_ui, nullptr, fs2_ui, Nu, 128, 1, 0);
    gemm(h2i, pW2_ui, nullptr, fd2_ui, Ni, 128, 1, 0);
    gemm(h2i, pW2_iu, nullptr, fs2_iu, Ni, 128, 1, 0);
    gemm(h2u, pW2_iu, nullptr, fd2_iu, Nu, 128, 1, 0);

    // ---- layer-2 attention scores (H=1, O=1) ----
    gat_attn_score<1, 1><<<cdiv(Nu, TB), TB, 0, stream>>>(fs2_ui, al2_ui, el2_ui, Nu);
    gat_attn_score<1, 1><<<cdiv(Ni, TB), TB, 0, stream>>>(fd2_ui, ar2_ui, er2_ui, Ni);
    gat_attn_score<1, 1><<<cdiv(Ni, TB), TB, 0, stream>>>(fs2_iu, al2_iu, el2_iu, Ni);
    gat_attn_score<1, 1><<<cdiv(Nu, TB), TB, 0, stream>>>(fd2_iu, ar2_iu, er2_iu, Nu);

    // ---- layer-2 softmax + aggregate : ui (dst = items) ----
    gat_fill<<<cdiv(Ni, TB), TB, 0, stream>>>(m2_ui, -INFINITY, (size_t)Ni);
    gat_fill<<<cdiv(Ni, TB), TB, 0, stream>>>(dn2_ui, 0.0f, (size_t)Ni);
    gat_fill<<<cdiv(Ni, TB), TB, 0, stream>>>(agg2_ui, 0.0f, (size_t)Ni);
    gat_edge_max<1><<<cdiv(E, TB), TB, 0, stream>>>(el2_ui, er2_ui, src_ui, dst_ui, e2_ui, m2_ui, E);
    gat_edge_exp<1><<<cdiv(E, TB), TB, 0, stream>>>(e2_ui, m2_ui, dst_ui, dn2_ui, E);
    gat_edge_aggr1<<<cdiv(E, TB), TB, 0, stream>>>(e2_ui, dn2_ui, fs2_ui, src_ui, dst_ui, agg2_ui, E);

    // ---- layer-2 softmax + aggregate : iu (dst = users) ----
    gat_fill<<<cdiv(Nu, TB), TB, 0, stream>>>(m2_iu, -INFINITY, (size_t)Nu);
    gat_fill<<<cdiv(Nu, TB), TB, 0, stream>>>(dn2_iu, 0.0f, (size_t)Nu);
    gat_fill<<<cdiv(Nu, TB), TB, 0, stream>>>(agg2_iu, 0.0f, (size_t)Nu);
    gat_edge_max<1><<<cdiv(E, TB), TB, 0, stream>>>(el2_iu, er2_iu, src_iu, dst_iu, e2_iu, m2_iu, E);
    gat_edge_exp<1><<<cdiv(E, TB), TB, 0, stream>>>(e2_iu, m2_iu, dst_iu, dn2_iu, E);
    gat_edge_aggr1<<<cdiv(E, TB), TB, 0, stream>>>(e2_iu, dn2_iu, fs2_iu, src_iu, dst_iu, agg2_iu, E);

    // ---- final output: concat([relu(user), relu(item)]) ----
    float* out = (float*)d_out;
    gat_out<<<cdiv(Nu, TB), TB, 0, stream>>>(agg2_iu, b2_iu, out, Nu);
    gat_out<<<cdiv(Ni, TB), TB, 0, stream>>>(agg2_ui, b2_ui, out + Nu, Ni);
}
